// AttentionConv_50843822850396
// MI455X (gfx1250) — matrix operand order repaired and checked by reading, not compiled, not run on hardware
//
#include <hip/hip_runtime.h>

// ---------------------------------------------------------------------------
// AttentionConv for MI455X (gfx1250, wave32, WMMA).
//   B=2, H=W=64, Cin=Cout=256, K_SZ=7, GROUPS=8.
// GEMMs run on v_wmma_f32_16x16x32_bf16 (f32 accum); softmax/gather on VALU.
// ---------------------------------------------------------------------------

typedef __attribute__((ext_vector_type(16))) __bf16 v16bf;
typedef __attribute__((ext_vector_type(8)))  __bf16 v8bf;
typedef __attribute__((ext_vector_type(8)))  float  v8f;

#define CIN   256
#define COUT  256
#define HH    64
#define WW    64
#define BB    2
#define PAD   3
#define HP    70            // 64 + 2*3
#define M_KV  (BB*HP*HP)    // 9800 padded rows
#define M_Q   (BB*HH*WW)    // 8192 rows

__device__ __forceinline__ unsigned short f2bf(float f) {
    union { float f; unsigned u; } x; x.f = f;
    unsigned r = x.u + 0x7FFFu + ((x.u >> 16) & 1u);   // round-to-nearest-even
    return (unsigned short)(r >> 16);
}

// ---- prep: W^T (bf16) for wq|wk|wv, contiguous --------------------------
__global__ void prep_weights(const float* __restrict__ wq,
                             const float* __restrict__ wk,
                             const float* __restrict__ wv,
                             unsigned short* __restrict__ wt) {
    int i = blockIdx.x * blockDim.x + threadIdx.x;      // 3*256*256 threads
    int wsel = i >> 16;
    int d = (i >> 8) & 255;   // output channel (row of W^T)
    int c = i & 255;          // input channel
    const float* W = (wsel == 0) ? wq : (wsel == 1) ? wk : wv;
    wt[i] = f2bf(W[c * COUT + d]);
}

// ---- prep: symmetric-padded x in bf16 -----------------------------------
__global__ void prep_xpad(const float* __restrict__ x,
                          unsigned short* __restrict__ xpad) {
    int i = blockIdx.x * blockDim.x + threadIdx.x;      // 2*70*70*256 threads
    int c = i & 255;
    int t = i >> 8;
    int j = t % HP;  t /= HP;
    int ii = t % HP; int b = t / HP;
    int h = ii - PAD; h = (h < 0) ? (-h - 1) : ((h > HH - 1) ? (2 * HH - 1 - h) : h);
    int w = j  - PAD; w = (w < 0) ? (-w - 1) : ((w > WW - 1) ? (2 * WW - 1 - w) : w);
    xpad[i] = f2bf(x[((b * HH + h) * WW + w) * CIN + c]);
}

// ---- fused K/V GEMM: [9800x256] x [256x256] twice, shared A frags -------
__global__ void gemm_kv(const __bf16* __restrict__ xpad,
                        const __bf16* __restrict__ wtk,
                        const __bf16* __restrict__ wtv,
                        float* __restrict__ kf, float* __restrict__ vf) {
    const int mtiles = (M_KV + 15) >> 4;                // 613
    int wave = (blockIdx.x * blockDim.x + threadIdx.x) >> 5;   // wave-uniform
    int lane = threadIdx.x & 31;
    if (wave >= mtiles * 16) return;                    // uniform: EXEC stays full
    int row0 = (wave >> 4) << 4;
    int col0 = (wave & 15) << 4;

    int arow = row0 + (lane & 15);
    if (arow > M_KV - 1) arow = M_KV - 1;               // clamp reads, keep EXEC=all-1
    int ksel = lane >> 4;                               // 0: K low half, 1: K high half

    const __bf16* aB  = xpad + (size_t)arow * CIN + ksel * 16;
    const __bf16* bkB = wtk  + (size_t)(col0 + (lane & 15)) * CIN + ksel * 16;
    const __bf16* bvB = wtv  + (size_t)(col0 + (lane & 15)) * CIN + ksel * 16;
    __builtin_prefetch(bkB, 0, 3);                      // -> global_prefetch_b8

    v8f ck = {}; v8f cv = {};
#pragma unroll
    for (int kk = 0; kk < CIN / 32; ++kk) {
        v8bf alo = *(const v8bf*)(aB + kk * 32);        // K {0..7}  (+16 if ksel)
        v8bf ahi = *(const v8bf*)(aB + kk * 32 + 8);    // K {8..15} (+16 if ksel)
        v16bf a = __builtin_shufflevector(alo, ahi,
                    0,1,2,3,4,5,6,7,8,9,10,11,12,13,14,15);
        v16bf bk = *(const v16bf*)(bkB + kk * 32);
        v16bf bv = *(const v16bf*)(bvB + kk * 32);
        ck = __builtin_amdgcn_wmma_f32_16x16x32_bf16(false, a, false, bk,
                                                     (short)0, ck, false, false);
        cv = __builtin_amdgcn_wmma_f32_16x16x32_bf16(false, a, false, bv,
                                                     (short)0, cv, false, false);
    }
    int drow0 = row0 + (ksel ? 8 : 0);
    int dcol  = col0 + (lane & 15);
#pragma unroll
    for (int v = 0; v < 8; ++v) {
        int r = drow0 + v;
        if (r < M_KV) {
            kf[(size_t)r * COUT + dcol] = ck[v];
            vf[(size_t)r * COUT + dcol] = cv[v];
        }
    }
}

// ---- Q GEMM: rows remapped through pad interior -------------------------
__global__ void gemm_q(const __bf16* __restrict__ xpad,
                       const __bf16* __restrict__ wtq,
                       float* __restrict__ q) {
    int wave = (blockIdx.x * blockDim.x + threadIdx.x) >> 5;
    int lane = threadIdx.x & 31;
    int row0 = (wave >> 4) << 4;                        // 512 row tiles exactly
    int col0 = (wave & 15) << 4;

    int r = row0 + (lane & 15);                         // r < 8192 always
    int b = r >> 12, rem = r & 4095, h = rem >> 6, w = rem & 63;
    int arow = b * (HP * HP) + (h + PAD) * HP + (w + PAD);
    int ksel = lane >> 4;

    const __bf16* aB = xpad + (size_t)arow * CIN + ksel * 16;
    const __bf16* bB = wtq  + (size_t)(col0 + (lane & 15)) * CIN + ksel * 16;

    v8f c = {};
#pragma unroll
    for (int kk = 0; kk < CIN / 32; ++kk) {
        v8bf alo = *(const v8bf*)(aB + kk * 32);
        v8bf ahi = *(const v8bf*)(aB + kk * 32 + 8);
        v16bf a = __builtin_shufflevector(alo, ahi,
                    0,1,2,3,4,5,6,7,8,9,10,11,12,13,14,15);
        v16bf bq = *(const v16bf*)(bB + kk * 32);
        c = __builtin_amdgcn_wmma_f32_16x16x32_bf16(false, a, false, bq,
                                                    (short)0, c, false, false);
    }
    int drow0 = row0 + (ksel ? 8 : 0);
    int dcol  = col0 + (lane & 15);
#pragma unroll
    for (int v = 0; v < 8; ++v)
        q[(size_t)(drow0 + v) * COUT + dcol] = c[v];
}

// ---- scrambled-reshape gather + online softmax + weighted sum -----------
__global__ void attn(const float* __restrict__ q,
                     const float* __restrict__ kf,
                     const float* __restrict__ vf,
                     const float* __restrict__ rh,
                     const float* __restrict__ rw,
                     float* __restrict__ out) {
    unsigned i = blockIdx.x * blockDim.x + threadIdx.x; // 2 * 2^20 threads
    unsigned b = i >> 20;                               // per-batch size = 2^20
    unsigned n = i & 0xFFFFFu;
    float qv = q[i];                                    // reshape is identity on flat
    const float* kfb = kf + (size_t)b * (HP * HP) * COUT;
    const float* vfb = vf + (size_t)b * (HP * HP) * COUT;

    unsigned f0 = n * 49u;
    float m = -__builtin_inff(), sum = 0.f, acc = 0.f;
    for (int k2 = 0; k2 < 49; ++k2) {
        unsigned f  = f0 + (unsigned)k2;                // flat into [H,W,7,7,C]
        unsigned c  = f & 255u;
        unsigned t  = f >> 8;
        unsigned kw = t % 7u;  unsigned t2 = t / 7u;
        unsigned kh = t2 % 7u; unsigned t3 = t2 / 7u;
        unsigned w  = t3 & 63u;
        unsigned h  = t3 >> 6;
        unsigned off = ((h + kh) * HP + (w + kw)) * COUT + c;
        float kvl = kfb[off];
        float bias = (c < 128u) ? rh[kh * 128u + c] : rw[kw * 128u + (c - 128u)];
        float vvl = vfb[off] + bias;
        float s  = qv * kvl;
        float nm = fmaxf(m, s);
        float cr = __expf(m - nm);
        float e  = __expf(s - nm);
        sum = sum * cr + e;
        acc = acc * cr + e * vvl;
        m = nm;
    }
    out[i] = acc / sum;
}

// ---------------------------------------------------------------------------
extern "C" void kernel_launch(void* const* d_in, const int* in_sizes, int n_in,
                              void* d_out, int out_size, void* d_ws, size_t ws_size,
                              hipStream_t stream) {
    const float* x  = (const float*)d_in[0];
    const float* wq = (const float*)d_in[1];
    const float* wk = (const float*)d_in[2];
    const float* wv = (const float*)d_in[3];
    const float* rh = (const float*)d_in[4];
    const float* rw = (const float*)d_in[5];
    float* out = (float*)d_out;

    // workspace layout (all 256B-aligned)
    char* ws = (char*)d_ws;
    unsigned short* wt   = (unsigned short*)ws;                 // 3*65536*2   = 393,216 B
    unsigned short* xpad = (unsigned short*)(ws + 393216);      // 2,508,800*2 = 5,017,600 B
    size_t o = 393216 + 5017600;
    float* qf  = (float*)(ws + o);               o += (size_t)M_Q  * COUT * 4; // 8,388,608
    float* kff = (float*)(ws + o);               o += (size_t)M_KV * COUT * 4; // 10,035,200
    float* vff = (float*)(ws + o);                                            // 10,035,200

    prep_weights<<<768, 256, 0, stream>>>(wq, wk, wv, wt);
    prep_xpad  <<<9800, 256, 0, stream>>>(x, xpad);

    const __bf16* wtq = (const __bf16*)wt;
    const __bf16* wtk = (const __bf16*)(wt + 65536);
    const __bf16* wtv = (const __bf16*)(wt + 131072);
    const __bf16* xpb = (const __bf16*)xpad;

    // 613*16 = 9808 tiles, 4 waves/block -> 2452 blocks
    gemm_kv<<<2452, 128, 0, stream>>>(xpb, wtk, wtv, kff, vff);
    // 512*16 = 8192 tiles -> 2048 blocks
    gemm_q <<<2048, 128, 0, stream>>>(xpb, wtq, qf);

    attn<<<8192, 256, 0, stream>>>(qf, kff, vff, rh, rw, out);
    (void)in_sizes; (void)n_in; (void)out_size; (void)ws_size;
}
